// GemmaTensorProductAttention_50706383896647
// MI455X (gfx1250) — matrix operand order repaired and checked by reading, not compiled, not run on hardware
//
#include <hip/hip_runtime.h>
#include <hip/hip_bf16.h>

// ---------------------------------------------------------------- constants
namespace cfg {
constexpr int Bb   = 2;
constexpr int Ss   = 2048;
constexpr int HID  = 3584;
constexpr int Hh   = 16;
constexpr int KVh  = 8;
constexpr int Dd   = 256;
constexpr int RQ   = 6;
constexpr int RK   = 2;
constexpr int RV   = 2;
constexpr int TOK  = Bb * Ss;              // 4096 tokens
constexpr int NCAT = Hh*RQ + KVh*RK + KVh*RV + RQ*Dd + RK*Dd + RV*Dd; // 2688
constexpr int HD   = Hh * Dd;              // 4096
constexpr float EPS     = 1e-6f;
constexpr float SOFTCAP = 50.0f;
constexpr float SCALE   = 0.0625f;         // 256^-0.5
constexpr float NEG     = -2.3819763e+38f;
}

typedef __bf16 bf16_t;
typedef __attribute__((ext_vector_type(16))) __bf16 v16bf;
typedef __attribute__((ext_vector_type(8)))  float  v8f;
typedef __attribute__((ext_vector_type(4)))  int    v4i;

// ---------------------------------------------------------------- async copy
#if defined(__has_builtin)
#  if __has_builtin(__builtin_amdgcn_global_load_async_to_lds_b128)
#    define ASYNC_LDS 1
#  endif
#endif
#ifndef ASYNC_LDS
#  define ASYNC_LDS 0
#endif

// Copy 16 bytes global -> LDS. Uses gfx1250 GLOBAL_LOAD_ASYNC_TO_LDS_B128
// (tracked by ASYNCcnt) when the builtin is available, else a b128 round trip.
// Builtin signature (from hipcc diagnostic): (v4i addrspace(1)*, v4i addrspace(3)*, imm offset, imm cpol).
__device__ __forceinline__ void cp_b128(void* lds, const void* g) {
#if ASYNC_LDS
  typedef __attribute__((address_space(1))) v4i* gp4_t;
  typedef __attribute__((address_space(3))) v4i* lp4_t;
  gp4_t gp = (gp4_t)(uintptr_t)g;
  lp4_t lp = (lp4_t)(uintptr_t)lds;
  __builtin_amdgcn_global_load_async_to_lds_b128(gp, lp, 0, 0);
#else
  *reinterpret_cast<uint4*>(lds) = *reinterpret_cast<const uint4*>(g);
#endif
}

__device__ __forceinline__ void cp_wait() {
#if ASYNC_LDS
  asm volatile("s_wait_asynccnt 0x0" ::: "memory");
#endif
}

// ---------------------------------------------------------------- WMMA helpers
__device__ __forceinline__ v8f wmma_bf16(v16bf a, v16bf b, v8f c) {
  // D(16x16,f32) = A(16x32,bf16) * B(32x16,bf16) + C
  return __builtin_amdgcn_wmma_f32_16x16x32_bf16(
      /*neg_a=*/false, a, /*neg_b=*/false, b,
      /*c_mod=*/(short)0, c, /*reuse_a=*/false, /*reuse_b=*/false);
}

__device__ __forceinline__ v8f vzero8() {
  v8f z;
#pragma unroll
  for (int i = 0; i < 8; ++i) z[i] = 0.0f;
  return z;
}

// A-matrix 16x32 bf16 fragment, row-major source with leading dim `lda`.
// Layout: lane L<16 -> row L, K = 0..15 ; lane L>=16 -> row L-16,
// K = 16..31.
__device__ __forceinline__ v16bf load_a16x32(const bf16_t* base, int lda) {
  const int lane = threadIdx.x & 31;
  const bf16_t* p = base + (size_t)(lane & 15) * lda + ((lane >> 4) << 4);
  v16bf f;
#pragma unroll
  for (int i = 0; i < 16; ++i) f[i] = p[i];
  return f;
}

// B-matrix 32x16 bf16 fragment from an "N-major" source: element (k, n) lives
// at base[n*ldn + k]. Layout used here (matches A): lane n (0..15) holds K rows 0..15 packed two
// per VGPR; lanes 16..31 hold K rows 16..31 for column n-16.
// -> each lane reads 16 contiguous bf16.
__device__ __forceinline__ v16bf load_b32x16_nmajor(const bf16_t* base, int ldn) {
  const int lane = threadIdx.x & 31;
  const bf16_t* p = base + (size_t)(lane & 15) * ldn + ((lane >> 4) << 4);
  v16bf f;
#pragma unroll
  for (int i = 0; i < 16; ++i) f[i] = p[i];
  return f;
}

// C/D 16x16 f32 fragment store: VGPR i: lanes 0-15 -> row i, lanes 16-31 ->
// row i+8; column = lane&15.
__device__ __forceinline__ void store_c16x16(float* base, int ldc, v8f c) {
  const int lane = threadIdx.x & 31;
  const int n = lane & 15, rb = (lane >> 4) << 3;
#pragma unroll
  for (int i = 0; i < 8; ++i) base[(size_t)(rb + i) * ldc + n] = c[i];
}

// ---------------------------------------------------------------- converts
__global__ void k_f32_to_bf16(const float* __restrict__ src,
                              bf16_t* __restrict__ dst, int n) {
  for (int i = blockIdx.x * blockDim.x + threadIdx.x; i < n;
       i += gridDim.x * blockDim.x)
    dst[i] = (bf16_t)src[i];
}

// o_proj [K=HD, N=HID] f32 -> N-major bf16 [HID, HD].
__global__ void k_transpose_bf16(const float* __restrict__ src,
                                 bf16_t* __restrict__ dst) {
  using namespace cfg;
  const int n = HID * HD;
  for (int i = blockIdx.x * blockDim.x + threadIdx.x; i < n;
       i += gridDim.x * blockDim.x) {
    const int c = i / HD, k = i % HD;      // dst[c][k] contiguous writes
    dst[i] = (bf16_t)src[(size_t)k * HID + c];
  }
}

// Pack six factor weights ([HID, *] row-major) into ONE N-major bf16 matrix
// [NCAT, HID]; column order: A_q | A_k | A_v | B_q | B_k | B_v.
__global__ void k_pack_w(const float* __restrict__ waq, const float* __restrict__ wak,
                         const float* __restrict__ wav, const float* __restrict__ wbq,
                         const float* __restrict__ wbk, const float* __restrict__ wbv,
                         bf16_t* __restrict__ wcatT) {
  using namespace cfg;
  const int n = NCAT * HID;
  for (int i = blockIdx.x * blockDim.x + threadIdx.x; i < n;
       i += gridDim.x * blockDim.x) {
    const int c = i / HID, k = i % HID;    // wcatT[c][k], contiguous writes
    float v;
    if      (c <   96) v = waq[(size_t)k *   96 +  c];
    else if (c <  112) v = wak[(size_t)k *   16 + (c -   96)];
    else if (c <  128) v = wav[(size_t)k *   16 + (c -  112)];
    else if (c < 1664) v = wbq[(size_t)k * 1536 + (c -  128)];
    else if (c < 2176) v = wbk[(size_t)k *  512 + (c - 1664)];
    else               v = wbv[(size_t)k *  512 + (c - 2176)];
    wcatT[i] = (bf16_t)v;
  }
}

// ---------------------------------------------------------------- WMMA GEMM
// C[M,N](f32) = A[M,K](bf16, row-major) * B(bf16, stored N-major as Bt[N,K]).
// M%128==0, N%128==0, K%32==0. Block tile 128x128, 8 waves, wave tile 32x64.
__global__ __launch_bounds__(256)
void k_gemm_bf16(const bf16_t* __restrict__ A, const bf16_t* __restrict__ Bt,
                 float* __restrict__ C, int M, int N, int K) {
  constexpr int BM = 128, BN = 128, BK = 32;
  alignas(16) __shared__ bf16_t sA[BM * BK];   // row-major [BM][BK]
  alignas(16) __shared__ bf16_t sB[BN * BK];   // N-major  [BN][BK]

  const int tid  = threadIdx.x;
  const int wave = tid >> 5;
  const int wm   = (wave & 3) * 32;            // 4 m-waves
  const int wn   = (wave >> 2) * 64;           // 2 n-waves
  const int bm   = blockIdx.x * BM;
  const int bn   = blockIdx.y * BN;

  const int srow  = tid >> 1;                  // 0..127 (both tiles are 128x32)
  const int shalf = (tid & 1) << 4;            // 0 or 16 elements

  v8f acc[2][4];
#pragma unroll
  for (int i = 0; i < 2; ++i)
#pragma unroll
    for (int j = 0; j < 4; ++j) acc[i][j] = vzero8();

  for (int k0 = 0; k0 < K; k0 += BK) {
    const bf16_t* ga = A  + (size_t)(bm + srow) * K + k0 + shalf;
    const bf16_t* gb = Bt + (size_t)(bn + srow) * K + k0 + shalf;
    bf16_t* la = sA + srow * BK + shalf;
    bf16_t* lb = sB + srow * BK + shalf;
    cp_b128(la,     ga);
    cp_b128(la + 8, ga + 8);
    cp_b128(lb,     gb);
    cp_b128(lb + 8, gb + 8);
    if (k0 + BK < K) {
      __builtin_prefetch(ga + BK, 0, 1);       // global_prefetch_b8
      __builtin_prefetch(gb + BK, 0, 1);
    }
    cp_wait();
    __syncthreads();

    v16bf af[2], bf[4];
#pragma unroll
    for (int mt = 0; mt < 2; ++mt)
      af[mt] = load_a16x32(sA + (wm + mt * 16) * BK, BK);
#pragma unroll
    for (int nt = 0; nt < 4; ++nt)
      bf[nt] = load_b32x16_nmajor(sB + (wn + nt * 16) * BK, BK);
#pragma unroll
    for (int mt = 0; mt < 2; ++mt)
#pragma unroll
      for (int nt = 0; nt < 4; ++nt)
        acc[mt][nt] = wmma_bf16(af[mt], bf[nt], acc[mt][nt]);
    __syncthreads();
  }

#pragma unroll
  for (int mt = 0; mt < 2; ++mt)
#pragma unroll
    for (int nt = 0; nt < 4; ++nt)
      store_c16x16(C + (size_t)(bm + wm + mt * 16) * N + (bn + wn + nt * 16),
                   N, acc[mt][nt]);
}

// ---------------------------------------------------------------- QKV build
__device__ __forceinline__ float block_rms256(float x, float* sh, int d) {
  sh[d] = x * x;
  __syncthreads();
  for (int o = 128; o > 0; o >>= 1) {
    if (d < o) sh[d] += sh[d + o];
    __syncthreads();
  }
  const float r = rsqrtf(sh[0] * (1.0f / 256.0f) + cfg::EPS);
  __syncthreads();
  return r;
}

__device__ __forceinline__ float block_rope256(float x, const float* fc,
                                               float* sh, int d) {
  sh[d] = x;
  __syncthreads();
  const float other = sh[d ^ 128];
  __syncthreads();
  const int j = d & 127;
  const float c = fc[2 * j], sn = fc[2 * j + 1];
  return (d < 128) ? (x * c - other * sn) : (x * c + other * sn);
}

// One block (256 threads) per token; thread d owns head-dim column d.
__global__ __launch_bounds__(256)
void k_qkv(const float* __restrict__ proj,   // [TOK, NCAT]
           const float* __restrict__ freqs,  // [S, 128, 2]
           const float* __restrict__ qw,     // [D]
           const float* __restrict__ kw,     // [D]
           bf16_t* __restrict__ qb,          // [B,H,S,D]   (SCALE folded in)
           bf16_t* __restrict__ kb,          // [B,KV,S,D]
           bf16_t* __restrict__ vtb) {       // [B,KV,D,S]  (transposed)
  using namespace cfg;
  __shared__ float sh[256];
  __shared__ float shA[128];

  const int token = blockIdx.x;
  const int b = token / Ss, s = token % Ss;
  const int d = threadIdx.x;
  const float* p  = proj + (size_t)token * NCAT;
  const float* fc = freqs + (size_t)s * 256;

  float bq[RQ], bkr[RK], bvr[RV];
#pragma unroll
  for (int r = 0; r < RQ; ++r) bq[r]  = p[128  + r * Dd + d];
#pragma unroll
  for (int r = 0; r < RK; ++r) bkr[r] = p[1664 + r * Dd + d];
#pragma unroll
  for (int r = 0; r < RV; ++r) bvr[r] = p[2176 + r * Dd + d];

  const float wq = 1.0f + qw[d];
  const float wk = 1.0f + kw[d];
#pragma unroll
  for (int r = 0; r < RQ; ++r) {
    const float rr = block_rms256(bq[r], sh, d);
    bq[r] = block_rope256(bq[r] * rr * wq, fc, sh, d);
  }
#pragma unroll
  for (int r = 0; r < RK; ++r) {
    const float rr = block_rms256(bkr[r], sh, d);
    bkr[r] = block_rope256(bkr[r] * rr * wk, fc, sh, d);
  }

  if (d < 128) shA[d] = p[d];   // A_q(96) | A_k(16) | A_v(16)
  __syncthreads();

#pragma unroll
  for (int h = 0; h < Hh; ++h) {
    float q = 0.0f;
#pragma unroll
    for (int r = 0; r < RQ; ++r) q += shA[h * RQ + r] * bq[r];
    q *= (SCALE / (float)RQ);
    qb[((size_t)(b * Hh + h) * Ss + s) * Dd + d] = (bf16_t)q;
  }
#pragma unroll
  for (int kv = 0; kv < KVh; ++kv) {
    float k = 0.0f, v = 0.0f;
#pragma unroll
    for (int r = 0; r < RK; ++r) k += shA[96  + kv * RK + r] * bkr[r];
#pragma unroll
    for (int r = 0; r < RV; ++r) v += shA[112 + kv * RV + r] * bvr[r];
    k *= (1.0f / (float)RK);
    v *= (1.0f / (float)RV);
    kb [((size_t)(b * KVh + kv) * Ss + s) * Dd + d] = (bf16_t)k;
    vtb[((size_t)(b * KVh + kv) * Dd + d) * Ss + s] = (bf16_t)v;
  }
}

// ---------------------------------------------------------------- attention
// V B-fragment: B[k,n] = V[kv0+k, d0+n] = vt[(d0+n)*S + kv0+k]; second 16-kv
// half is clamped at the causal edge (its P columns are exactly zero there).
__device__ __forceinline__ v16bf load_v_frag(const bf16_t* vh, int kv0, int dt) {
  using namespace cfg;
  const int lane = threadIdx.x & 31;
  const int n = lane & 15;
  int kcol = kv0 + ((lane >> 4) << 4);
  if (kcol + 16 > Ss) kcol = Ss - 16;
  const bf16_t* p = vh + (size_t)(dt * 16 + n) * Ss + kcol;
  v16bf f;
#pragma unroll
  for (int i = 0; i < 16; ++i) f[i] = p[i];
  return f;
}

// One wave (32 threads) per (b, h, 16-row q tile). Flash-style online softmax
// with tanh softcap; P transposed through private LDS into A-fragment layout.
__global__ __launch_bounds__(32)
void k_attn(const bf16_t* __restrict__ qb, const bf16_t* __restrict__ kbuf,
            const bf16_t* __restrict__ vtb, bf16_t* __restrict__ attn_out) {
  using namespace cfg;
  __shared__ bf16_t pl[16 * 32];
  __builtin_amdgcn_s_wait_tensorcnt(0);

  const int bid = blockIdx.x;
  const int qt = bid & 127, h = (bid >> 7) & 15, b = bid >> 11;
  const int q0 = qt * 16;
  const int kvh = h >> 1;                                  // GRP = 2
  const int lane = threadIdx.x;
  const int col_n = lane & 15;
  const int row_off = (lane >> 4) << 3;

  const bf16_t* qh = qb   + (size_t)(b * Hh  + h)   * Ss * Dd;
  const bf16_t* kh = kbuf + (size_t)(b * KVh + kvh) * Ss * Dd;
  const bf16_t* vh = vtb  + (size_t)(b * KVh + kvh) * Dd * Ss;

  v16bf aq[8];
#pragma unroll
  for (int dk = 0; dk < 8; ++dk)
    aq[dk] = load_a16x32(qh + (size_t)q0 * Dd + dk * 32, Dd);

  v8f o[16];
#pragma unroll
  for (int dt = 0; dt < 16; ++dt) o[dt] = vzero8();
  float m[8], l[8];
#pragma unroll
  for (int i = 0; i < 8; ++i) { m[i] = -3.0e38f; l[i] = 0.0f; }

  for (int kv0 = 0; kv0 < q0 + 16; kv0 += 32) {
    // ---- scores: two 16x16 tiles, K-dim chained over D=256
    v8f s0 = vzero8(), s1 = vzero8();
    const int kv1 = (kv0 + 32 <= Ss) ? kv0 + 16 : Ss - 16;  // clamped tile-1
#pragma unroll
    for (int dk = 0; dk < 8; ++dk) {
      s0 = wmma_bf16(aq[dk],
                     load_b32x16_nmajor(kh + (size_t)kv0 * Dd + dk * 32, Dd), s0);
      s1 = wmma_bf16(aq[dk],
                     load_b32x16_nmajor(kh + (size_t)kv1 * Dd + dk * 32, Dd), s1);
    }

    // ---- softcap + causal mask + online softmax
    float p0[8], p1[8];
#pragma unroll
    for (int i = 0; i < 8; ++i) {
      float x0 = SOFTCAP * tanhf(s0[i] * (1.0f / SOFTCAP));
      float x1 = SOFTCAP * tanhf(s1[i] * (1.0f / SOFTCAP));
      const int row = q0 + row_off + i;
      if (kv0 + col_n      > row) x0 = NEG;
      if (kv0 + 16 + col_n > row) x1 = NEG;
      float r = fmaxf(x0, x1);
      r = fmaxf(r, __shfl_xor(r, 1));
      r = fmaxf(r, __shfl_xor(r, 2));
      r = fmaxf(r, __shfl_xor(r, 4));
      r = fmaxf(r, __shfl_xor(r, 8));   // reduce within the 16-lane half
      const float mn = fmaxf(m[i], r);
      const float alpha = __expf(m[i] - mn);
      const float e0 = __expf(x0 - mn);
      const float e1 = __expf(x1 - mn);
      float rs = e0 + e1;
      rs += __shfl_xor(rs, 1);
      rs += __shfl_xor(rs, 2);
      rs += __shfl_xor(rs, 4);
      rs += __shfl_xor(rs, 8);
      l[i] = l[i] * alpha + rs;
      m[i] = mn;
#pragma unroll
      for (int dt = 0; dt < 16; ++dt) o[dt][i] *= alpha;
      p0[i] = e0; p1[i] = e1;
    }

    // ---- transpose P (C layout -> A layout) through LDS
#pragma unroll
    for (int i = 0; i < 8; ++i) {
      const int r = row_off + i;
      pl[r * 32 + col_n]      = (bf16_t)p0[i];
      pl[r * 32 + 16 + col_n] = (bf16_t)p1[i];
    }
    __syncthreads();                      // single-wave WG: ordering only
    const v16bf pa = load_a16x32(pl, 32);

    // ---- O += P * V
#pragma unroll
    for (int dt = 0; dt < 16; ++dt)
      o[dt] = wmma_bf16(pa, load_v_frag(vh, kv0, dt), o[dt]);
    __syncthreads();
  }

  float inv[8];
#pragma unroll
  for (int i = 0; i < 8; ++i) inv[i] = 1.0f / l[i];
  bf16_t* outp = attn_out + (size_t)(b * Ss + q0) * HD + h * Dd;
#pragma unroll
  for (int dt = 0; dt < 16; ++dt)
#pragma unroll
    for (int i = 0; i < 8; ++i)
      outp[(size_t)(row_off + i) * HD + dt * 16 + col_n] =
          (bf16_t)(o[dt][i] * inv[i]);
}

// ---------------------------------------------------------------- launcher
extern "C" void kernel_launch(void* const* d_in, const int* in_sizes, int n_in,
                              void* d_out, int out_size, void* d_ws, size_t ws_size,
                              hipStream_t stream) {
  using namespace cfg;
  const float* hs    = (const float*)d_in[0];
  const float* freqs = (const float*)d_in[1];
  // d_in[2..7]: kv_write_indices (arange), caches (overwritten), mask (causal) — unused
  const float* waq = (const float*)d_in[8];
  const float* wak = (const float*)d_in[9];
  const float* wav = (const float*)d_in[10];
  const float* wbq = (const float*)d_in[11];
  const float* wbk = (const float*)d_in[12];
  const float* wbv = (const float*)d_in[13];
  const float* wo  = (const float*)d_in[14];
  const float* qw  = (const float*)d_in[15];
  const float* kw  = (const float*)d_in[16];
  float* out = (float*)d_out;

  char* w = (char*)d_ws;
  auto alloc = [&](size_t bytes) {
    char* p = w;
    w += (bytes + 255) & ~(size_t)255;
    return p;
  };
  bf16_t* hsb   = (bf16_t*)alloc((size_t)TOK * HID * 2);
  bf16_t* wcatT = (bf16_t*)alloc((size_t)NCAT * HID * 2);  // N-major [NCAT,HID]
  float*  proj  = (float*) alloc((size_t)TOK * NCAT * 4);
  bf16_t* qb    = (bf16_t*)alloc((size_t)Bb * Hh  * Ss * Dd * 2);
  bf16_t* kb    = (bf16_t*)alloc((size_t)Bb * KVh * Ss * Dd * 2);
  bf16_t* vtb   = (bf16_t*)alloc((size_t)Bb * KVh * Dd * Ss * 2);
  bf16_t* attn  = (bf16_t*)alloc((size_t)TOK * HD * 2);
  bf16_t* woT   = (bf16_t*)alloc((size_t)HID * HD * 2);    // N-major [HID,HD]

  // 0) converts / packing (weights go to N-major bf16 for vectorized staging)
  k_f32_to_bf16<<<4096, 256, 0, stream>>>(hs, hsb, TOK * HID);
  k_pack_w<<<4096, 256, 0, stream>>>(waq, wak, wav, wbq, wbk, wbv, wcatT);
  k_transpose_bf16<<<4096, 256, 0, stream>>>(wo, woT);

  // 1) fused factor projection: [4096,3584] x [3584,2688] -> proj (f32)
  k_gemm_bf16<<<dim3(TOK / 128, NCAT / 128), 256, 0, stream>>>(
      hsb, wcatT, proj, TOK, NCAT, HID);

  // 2) RMSNorm + RoPE + rank contraction -> Q/K/V (bf16)
  k_qkv<<<TOK, 256, 0, stream>>>(proj, freqs, qw, kw, qb, kb, vtb);

  // 3) causal softcapped attention
  k_attn<<<Bb * Hh * (Ss / 16), 32, 0, stream>>>(qb, kb, vtb, attn);

  // 4) output projection: [4096,4096] x [4096,3584] -> d_out (f32)
  k_gemm_bf16<<<dim3(TOK / 128, HID / 128), 256, 0, stream>>>(
      attn, woT, out, TOK, HID, HD);
}
